// Pct_seg_7687991460140
// MI455X (gfx1250) — matrix operand order repaired and checked by reading, not compiled, not run on hardware
//
#include <hip/hip_runtime.h>
#include <hip/hip_bf16.h>

#define NPTS 8192

typedef __attribute__((ext_vector_type(16))) __bf16 v16bf;
typedef __attribute__((ext_vector_type(8)))  float  v8f;
typedef __attribute__((ext_vector_type(8)))  __bf16 bf8v;

#define ACT_NONE 0
#define ACT_RELU 1
#define ACT_LRELU 2
#define OUT_FM_BF16 0
#define OUT_CM_BF16 1
#define OUT_CM_F32 2

// ---------------- WMMA fragment helpers (wave32, 16x16x32 bf16) ----------------
// A (16x32, row-major MxK): lane holds row m0+(l&15); halves h -> K = kb+16*hi+h
// (hi = lane>>4, contiguous 32B).  Two 16B chunks per lane.
__device__ __forceinline__ v16bf frag_a(const __bf16* __restrict__ A, int lda,
                                        int row0, int kb, int lane) {
  int r  = row0 + (lane & 15);
  int hi = lane >> 4;
  const __bf16* p = A + (size_t)r * lda + kb + 16 * hi;
  bf8v c0 = *(const bf8v*)p;
  bf8v c1 = *(const bf8v*)(p + 8);
  v16bf f;
#pragma unroll
  for (int i = 0; i < 8; ++i) { f[i] = c0[i]; f[i + 8] = c1[i]; }
  return f;
}

// B (32x16, KxN) read from a feature-major activation act[n][k]:
// lane holds column n0+(l&15); halves h -> K = kb + 16*hi + h (contiguous 32B).
__device__ __forceinline__ v16bf frag_b(const __bf16* __restrict__ Bfm, int ldb,
                                        int col0, int kb, int lane) {
  int n  = col0 + (lane & 15);
  int hi = lane >> 4;
  const __bf16* p = Bfm + (size_t)n * ldb + kb + 16 * hi;
  bf8v c0 = *(const bf8v*)p;
  bf8v c1 = *(const bf8v*)(p + 8);
  v16bf f;
#pragma unroll
  for (int i = 0; i < 8; ++i) { f[i] = c0[i]; f[i + 8] = c1[i]; }
  return f;
}

__device__ __forceinline__ v8f wmma_bf(v16bf a, v16bf b, v8f c) {
  return __builtin_amdgcn_wmma_f32_16x16x32_bf16(false, a, false, b, (short)0, c,
                                                 false, false);
}

__device__ __forceinline__ v8f zero8() {
  v8f z;
#pragma unroll
  for (int i = 0; i < 8; ++i) z[i] = 0.f;
  return z;
}

// ---------------- Generic fused GEMM: Y = act(W'X + bias') (+resid) ----------------
// W' bf16 MxKp row-major (BN scale folded), X feature-major [n][k] bf16.
// Block = 128 threads (4 waves, 2m x 2n), block tile 64m x 128n, wave tile 32m x 64n.
__global__ __launch_bounds__(128) void gemm_fm(
    const __bf16* __restrict__ A, int lda, const __bf16* __restrict__ B, int ldb,
    const float* __restrict__ bias, const __bf16* __restrict__ resid, int ldr,
    void* __restrict__ out, int ldo, int K, int act, int outmode) {
  int lane = threadIdx.x & 31;
  int w = threadIdx.x >> 5;
  int m0 = blockIdx.y * 64 + (w & 1) * 32;
  int n0 = blockIdx.x * 128 + (w >> 1) * 64;

  v8f acc[2][4];
#pragma unroll
  for (int i = 0; i < 2; ++i)
#pragma unroll
    for (int j = 0; j < 4; ++j) acc[i][j] = zero8();

  for (int kb = 0; kb < K; kb += 32) {
    v16bf a0 = frag_a(A, lda, m0, kb, lane);
    v16bf a1 = frag_a(A, lda, m0 + 16, kb, lane);
#pragma unroll
    for (int j = 0; j < 4; ++j) {
      v16bf b = frag_b(B, ldb, n0 + j * 16, kb, lane);
      acc[0][j] = wmma_bf(a0, b, acc[0][j]);
      acc[1][j] = wmma_bf(a1, b, acc[1][j]);
    }
  }

  int hi = lane >> 4, lo = lane & 15;
#pragma unroll
  for (int i = 0; i < 2; ++i) {
    int mt = m0 + i * 16 + 8 * hi;  // 8 consecutive output channels
#pragma unroll
    for (int j = 0; j < 4; ++j) {
      int nt = n0 + j * 16 + lo;  // point index
      float vals[8];
#pragma unroll
      for (int r = 0; r < 8; ++r) {
        float v = acc[i][j][r] + bias[mt + r];
        if (act == ACT_RELU) v = fmaxf(v, 0.f);
        else if (act == ACT_LRELU) v = (v > 0.f) ? v : 0.2f * v;
        vals[r] = v;
      }
      if (resid) {
        bf8v rv = *(const bf8v*)(resid + (size_t)nt * ldr + mt);
#pragma unroll
        for (int r = 0; r < 8; ++r) vals[r] += (float)rv[r];
      }
      if (outmode == OUT_FM_BF16) {
        bf8v ov;
#pragma unroll
        for (int r = 0; r < 8; ++r) ov[r] = (__bf16)vals[r];
        *(bf8v*)((__bf16*)out + (size_t)nt * ldo + mt) = ov;
      } else if (outmode == OUT_CM_BF16) {
        __bf16* o = (__bf16*)out;
#pragma unroll
        for (int r = 0; r < 8; ++r) o[(size_t)(mt + r) * NPTS + nt] = (__bf16)vals[r];
      } else {
        float* o = (float*)out;
#pragma unroll
        for (int r = 0; r < 8; ++r) o[(size_t)(mt + r) * NPTS + nt] = vals[r];
      }
    }
  }
}

// ---------------- Attention pass 1: per-row online-softmax stats ----------------
// energy[n][m] = sum_d qk[n][d]*qk[m][d]; store rowmax and 1/rowsum.
__global__ __launch_bounds__(128) void attn_rowstats(const __bf16* __restrict__ qk,
                                                     float* __restrict__ rmax,
                                                     float* __restrict__ rinv) {
  int lane = threadIdx.x & 31;
  int w = threadIdx.x >> 5;
  int n0 = blockIdx.x * 64 + w * 16;
  v16bf a0 = frag_a(qk, 64, n0, 0, lane);
  v16bf a1 = frag_a(qk, 64, n0, 32, lane);
  float mrow[8], srow[8];
#pragma unroll
  for (int r = 0; r < 8; ++r) { mrow[r] = -3.0e38f; srow[r] = 0.f; }

  for (int m0 = 0; m0 < NPTS; m0 += 16) {
    v16bf b0 = frag_b(qk, 64, m0, 0, lane);
    v16bf b1 = frag_b(qk, 64, m0, 32, lane);
    v8f c = zero8();
    c = wmma_bf(a0, b0, c);
    c = wmma_bf(a1, b1, c);
#pragma unroll
    for (int r = 0; r < 8; ++r) {
      float v = c[r];
      if (v > mrow[r]) {
        srow[r] = srow[r] * __expf(mrow[r] - v) + 1.f;
        mrow[r] = v;
      } else {
        srow[r] += __expf(v - mrow[r]);
      }
    }
  }
  // combine across the 16 lanes that share the same 8 rows
#pragma unroll
  for (int off = 1; off < 16; off <<= 1) {
#pragma unroll
    for (int r = 0; r < 8; ++r) {
      float om = __shfl_xor(mrow[r], off, 32);
      float os = __shfl_xor(srow[r], off, 32);
      float M = fmaxf(mrow[r], om);
      srow[r] = srow[r] * __expf(mrow[r] - M) + os * __expf(om - M);
      mrow[r] = M;
    }
  }
  if ((lane & 15) == 0) {
    int base = n0 + 8 * (lane >> 4);
#pragma unroll
    for (int r = 0; r < 8; ++r) {
      rmax[base + r] = mrow[r];
      rinv[base + r] = 1.f / srow[r];
    }
  }
}

// ---------------- Attention pass 2: column sums of softmax(att) ----------------
__global__ __launch_bounds__(128) void attn_colsum(const __bf16* __restrict__ qk,
                                                   const float* __restrict__ rmax,
                                                   const float* __restrict__ rinv,
                                                   float* __restrict__ colsum) {
  int lane = threadIdx.x & 31;
  int w = threadIdx.x >> 5;
  int m0 = blockIdx.x * 64 + w * 16;
  v16bf b0 = frag_b(qk, 64, m0, 0, lane);
  v16bf b1 = frag_b(qk, 64, m0, 32, lane);
  float cs = 0.f;
  for (int n0 = 0; n0 < NPTS; n0 += 16) {
    v16bf a0 = frag_a(qk, 64, n0, 0, lane);
    v16bf a1 = frag_a(qk, 64, n0, 32, lane);
    v8f c = zero8();
    c = wmma_bf(a0, b0, c);
    c = wmma_bf(a1, b1, c);
    int rb = n0 + 8 * (lane >> 4);
#pragma unroll
    for (int r = 0; r < 8; ++r) cs += __expf(c[r] - rmax[rb + r]) * rinv[rb + r];
  }
  cs += __shfl_xor(cs, 16, 32);
  if (lane < 16) colsum[m0 + lane] = cs;
}

// ---------------- Attention pass 3: s = h - (xv x att)/(1e-9+colsum) ----------------
// Block: 64 output points x all 256 channels, 8 waves (256 threads).
// Per 64-wide n strip: 16 energy tiles produced cooperatively (2 per wave, ni=w>>1,
// mi = (w&1)*2 + {0,1}), exponentiated into LDS in B-fragment layout, then each wave
// consumes them against its 32-channel strip of xv (A-frags reused across 4 m-tiles).
__global__ __launch_bounds__(256) void attn_apply(
    const __bf16* __restrict__ qk, const __bf16* __restrict__ xv /*[256][N]*/,
    const __bf16* __restrict__ p2 /*[N][256]*/, const float* __restrict__ rmax,
    const float* __restrict__ rinv, const float* __restrict__ colsum,
    __bf16* __restrict__ s /*[N][256]*/) {
  __shared__ __bf16 SA[64 * 72];  // [m_local][k_local], padded stride 72
  int lane = threadIdx.x & 31;
  int w = threadIdx.x >> 5;
  int hi = lane >> 4, lo = lane & 15;
  int m0 = blockIdx.x * 64;
  int cbase = w * 32;        // this wave's channel strip
  int ni = w >> 1;           // this wave's n-tile within the 64-strip
  int miA = (w & 1) * 2;     // this wave's two m-tiles for energy production

  v16bf eb[2][2];  // [j][kb] B-frags of energy GEMM for columns m0+ (miA+j)*16
#pragma unroll
  for (int j = 0; j < 2; ++j) {
    eb[j][0] = frag_b(qk, 64, m0 + (miA + j) * 16, 0, lane);
    eb[j][1] = frag_b(qk, 64, m0 + (miA + j) * 16, 32, lane);
  }

  v8f acc[2][4];  // [c-tile][m-tile]
#pragma unroll
  for (int i = 0; i < 2; ++i)
#pragma unroll
    for (int j = 0; j < 4; ++j) acc[i][j] = zero8();

  for (int n0 = 0; n0 < NPTS; n0 += 64) {
    int nw = n0 + ni * 16;
    v16bf ea0 = frag_a(qk, 64, nw, 0, lane);
    v16bf ea1 = frag_a(qk, 64, nw, 32, lane);
    int rb = nw + 8 * hi;
    float rm[8], ri[8];
#pragma unroll
    for (int r = 0; r < 8; ++r) { rm[r] = rmax[rb + r]; ri[r] = rinv[rb + r]; }
#pragma unroll
    for (int j = 0; j < 2; ++j) {
      v8f e = zero8();
      e = wmma_bf(ea0, eb[j][0], e);
      e = wmma_bf(ea1, eb[j][1], e);
      bf8v st;
#pragma unroll
      for (int r = 0; r < 8; ++r) st[r] = (__bf16)(__expf(e[r] - rm[r]) * ri[r]);
      *(bf8v*)&SA[((miA + j) * 16 + lo) * 72 + ni * 16 + 8 * hi] = st;
    }
    __syncthreads();
#pragma unroll
    for (int kk = 0; kk < 64; kk += 32) {
      v16bf af0 = frag_a(xv, NPTS, cbase, n0 + kk, lane);
      v16bf af1 = frag_a(xv, NPTS, cbase + 16, n0 + kk, lane);
#pragma unroll
      for (int mi = 0; mi < 4; ++mi) {
        const __bf16* p = &SA[(mi * 16 + lo) * 72 + kk + 16 * hi];
        bf8v c0 = *(const bf8v*)p;
        bf8v c1 = *(const bf8v*)(p + 8);
        v16bf bfr;
#pragma unroll
        for (int i = 0; i < 8; ++i) { bfr[i] = c0[i]; bfr[i + 8] = c1[i]; }
        acc[0][mi] = wmma_bf(af0, bfr, acc[0][mi]);
        acc[1][mi] = wmma_bf(af1, bfr, acc[1][mi]);
      }
    }
    __syncthreads();
  }

#pragma unroll
  for (int mi = 0; mi < 4; ++mi) {
    int pt = m0 + mi * 16 + lo;
    float inv = 1.f / (1e-9f + colsum[pt]);
#pragma unroll
    for (int t = 0; t < 2; ++t) {
      int cb = cbase + t * 16 + 8 * hi;
      bf8v hv = *(const bf8v*)&p2[(size_t)pt * 256 + cb];
      bf8v ov;
#pragma unroll
      for (int r = 0; r < 8; ++r) ov[r] = (__bf16)((float)hv[r] - acc[t][mi][r] * inv);
      *(bf8v*)&s[(size_t)pt * 256 + cb] = ov;
    }
  }
}

// ---------------- Small prep kernels ----------------
// Fold BN (and optional linear bias) into weights: A = sc*W, bias = beta - sc*mean + sc*lb
__global__ void fold_weights(const float* __restrict__ W, const float* __restrict__ g,
                             const float* __restrict__ b, const float* __restrict__ m,
                             const float* __restrict__ v, const float* __restrict__ lb,
                             __bf16* __restrict__ A, float* __restrict__ bias, int M,
                             int K, int Kp) {
  int id = blockIdx.x * blockDim.x + threadIdx.x;
  if (id >= M * Kp) return;
  int o = id / Kp, k = id % Kp;
  float sc = g ? g[o] * rsqrtf(v[o] + 1e-5f) : 1.f;
  A[id] = (k < K) ? (__bf16)(sc * W[o * K + k]) : (__bf16)0.f;
  if (k == 0) {
    float bb = b ? (b[o] - sc * m[o]) : 0.f;
    if (lb) bb += sc * lb[o];
    bias[o] = bb;
  }
}

// x (N x 628 fp32) -> feature-major bf16 padded to 640
__global__ void prep_input(const float* __restrict__ x, __bf16* __restrict__ B0) {
  int id = blockIdx.x * blockDim.x + threadIdx.x;
  if (id >= NPTS * 640) return;
  int n = id / 640, k = id % 640;
  B0[id] = (k < 628) ? (__bf16)x[n * 628 + k] : (__bf16)0.f;
}

// B4[n][k]: k<128 -> f[n][k]; else neighbor j=(k-128)/128 channel c
__global__ void gather_neigh(const __bf16* __restrict__ f, const int* __restrict__ idx,
                             __bf16* __restrict__ B4) {
  int id = blockIdx.x * blockDim.x + threadIdx.x;
  if (id >= NPTS * 512) return;
  int n = id / 512, k = id % 512;
  if (k < 128) {
    B4[id] = f[n * 128 + k];
  } else {
    int j = (k - 128) >> 7, c = (k - 128) & 127;
    int nn = idx[n * 3 + j];
    B4[id] = f[nn * 128 + c];
  }
}

// ---------------- Host orchestration ----------------
extern "C" void kernel_launch(void* const* d_in, const int* in_sizes, int n_in,
                              void* d_out, int out_size, void* d_ws, size_t ws_size,
                              hipStream_t stream) {
  (void)in_sizes; (void)n_in; (void)out_size; (void)ws_size;
  const float* x = (const float*)d_in[0];
  const int* index = (const int*)d_in[50];

  struct Layer { int M, K, Kp; const float *W, *g, *b, *m, *v, *lb; };
  Layer L[11] = {
      {512, 628, 640, (const float*)d_in[1],  (const float*)d_in[2],  (const float*)d_in[3],  (const float*)d_in[4],  (const float*)d_in[5],  nullptr},
      {256, 512, 512, (const float*)d_in[6],  (const float*)d_in[7],  (const float*)d_in[8],  (const float*)d_in[9],  (const float*)d_in[10], nullptr},
      {128, 256, 256, (const float*)d_in[11], (const float*)d_in[12], (const float*)d_in[13], (const float*)d_in[14], (const float*)d_in[15], nullptr},
      {256, 512, 512, (const float*)d_in[16], (const float*)d_in[17], (const float*)d_in[18], (const float*)d_in[19], (const float*)d_in[20], nullptr},
      {256, 256, 256, (const float*)d_in[21], (const float*)d_in[22], (const float*)d_in[23], (const float*)d_in[24], (const float*)d_in[25], nullptr},
      {256, 256, 256, (const float*)d_in[26], (const float*)d_in[27], (const float*)d_in[28], (const float*)d_in[29], (const float*)d_in[30], nullptr},
      {64, 256, 256,  (const float*)d_in[31], nullptr, nullptr, nullptr, nullptr, nullptr},                       // wqk
      {256, 256, 256, (const float*)d_in[32], nullptr, nullptr, nullptr, nullptr, (const float*)d_in[33]},        // wv + bv
      {256, 256, 256, (const float*)d_in[34], (const float*)d_in[36], (const float*)d_in[37], (const float*)d_in[38], (const float*)d_in[39], (const float*)d_in[35]},  // wt + bt + sabn
      {512, 512, 512, (const float*)d_in[40], (const float*)d_in[41], (const float*)d_in[42], (const float*)d_in[43], (const float*)d_in[44], nullptr},
      {256, 512, 512, (const float*)d_in[45], (const float*)d_in[46], (const float*)d_in[47], (const float*)d_in[48], (const float*)d_in[49], nullptr},
  };

  char* base = (char*)d_ws;
  size_t off = 0;
  auto alloc = [&](size_t bytes) -> char* {
    off = (off + 255) & ~(size_t)255;
    char* p = base + off;
    off += bytes;
    return p;
  };

  __bf16* wA[11];
  float* wb[11];
  for (int i = 0; i < 11; ++i) {
    wA[i] = (__bf16*)alloc((size_t)L[i].M * L[i].Kp * 2);
    wb[i] = (float*)alloc((size_t)L[i].M * 4);
  }
  __bf16* B0    = (__bf16*)alloc((size_t)NPTS * 640 * 2);
  __bf16* act1  = (__bf16*)alloc((size_t)NPTS * 512 * 2);
  __bf16* act2  = (__bf16*)alloc((size_t)NPTS * 256 * 2);
  __bf16* act3  = (__bf16*)alloc((size_t)NPTS * 128 * 2);
  __bf16* B4    = (__bf16*)alloc((size_t)NPTS * 512 * 2);
  __bf16* fuse  = (__bf16*)alloc((size_t)NPTS * 512 * 2);  // [hsum(0..255) | before(256..511)]
  __bf16* p1    = (__bf16*)alloc((size_t)NPTS * 256 * 2);
  __bf16* p2    = (__bf16*)alloc((size_t)NPTS * 256 * 2);
  __bf16* qk    = (__bf16*)alloc((size_t)NPTS * 64 * 2);
  __bf16* xv    = (__bf16*)alloc((size_t)256 * NPTS * 2);  // channel-major
  __bf16* sbuf  = (__bf16*)alloc((size_t)NPTS * 256 * 2);
  __bf16* fuse1 = (__bf16*)alloc((size_t)NPTS * 512 * 2);
  float* rmax   = (float*)alloc(NPTS * 4);
  float* rinv   = (float*)alloc(NPTS * 4);
  float* csum   = (float*)alloc(NPTS * 4);

  for (int i = 0; i < 11; ++i) {
    int total = L[i].M * L[i].Kp;
    fold_weights<<<(total + 255) / 256, 256, 0, stream>>>(
        L[i].W, L[i].g, L[i].b, L[i].m, L[i].v, L[i].lb, wA[i], wb[i], L[i].M,
        L[i].K, L[i].Kp);
  }
  prep_input<<<(NPTS * 640 + 255) / 256, 256, 0, stream>>>(x, B0);

  dim3 blk(128);
  int gx = NPTS / 128;  // 64 blocks in n
  // L0..L2: feature MLP
  gemm_fm<<<dim3(gx, 8), blk, 0, stream>>>(wA[0], 640, B0, 640, wb[0], nullptr, 0,
                                           act1, 512, 640, ACT_RELU, OUT_FM_BF16);
  gemm_fm<<<dim3(gx, 4), blk, 0, stream>>>(wA[1], 512, act1, 512, wb[1], nullptr, 0,
                                           act2, 256, 512, ACT_RELU, OUT_FM_BF16);
  gemm_fm<<<dim3(gx, 2), blk, 0, stream>>>(wA[2], 256, act2, 256, wb[2], nullptr, 0,
                                           act3, 128, 256, ACT_RELU, OUT_FM_BF16);
  gather_neigh<<<(NPTS * 512 + 255) / 256, 256, 0, stream>>>(act3, index, B4);
  // L3: -> 'before' stored in fuse[.,256:512]
  gemm_fm<<<dim3(gx, 4), blk, 0, stream>>>(wA[3], 512, B4, 512, wb[3], nullptr, 0,
                                           fuse + 256, 512, 512, ACT_RELU, OUT_FM_BF16);
  // Point_Transformer_Last pre-convs
  gemm_fm<<<dim3(gx, 4), blk, 0, stream>>>(wA[4], 256, fuse + 256, 512, wb[4], nullptr,
                                           0, p1, 256, 256, ACT_RELU, OUT_FM_BF16);
  gemm_fm<<<dim3(gx, 4), blk, 0, stream>>>(wA[5], 256, p1, 256, wb[5], nullptr, 0, p2,
                                           256, 256, ACT_RELU, OUT_FM_BF16);
  // q/k shared projection and v
  gemm_fm<<<dim3(gx, 1), blk, 0, stream>>>(wA[6], 256, p2, 256, wb[6], nullptr, 0, qk,
                                           64, 256, ACT_NONE, OUT_FM_BF16);
  gemm_fm<<<dim3(gx, 4), blk, 0, stream>>>(wA[7], 256, p2, 256, wb[7], nullptr, 0, xv,
                                           NPTS, 256, ACT_NONE, OUT_CM_BF16);
  // offset attention (flash-style, N x N never materialized)
  attn_rowstats<<<NPTS / 64, blk, 0, stream>>>(qk, rmax, rinv);
  attn_colsum<<<NPTS / 64, blk, 0, stream>>>(qk, rmax, rinv, csum);
  attn_apply<<<NPTS / 64, dim3(256), 0, stream>>>(qk, xv, p2, rmax, rinv, csum, sbuf);
  // wt conv + BN + relu, residual add h -> fuse[.,0:256]
  gemm_fm<<<dim3(gx, 4), blk, 0, stream>>>(wA[8], 256, sbuf, 256, wb[8], p2, 256, fuse,
                                           512, 256, ACT_RELU, OUT_FM_BF16);
  // conv_fuse
  gemm_fm<<<dim3(gx, 8), blk, 0, stream>>>(wA[9], 512, fuse, 512, wb[9], nullptr, 0,
                                           fuse1, 512, 512, ACT_LRELU, OUT_FM_BF16);
  gemm_fm<<<dim3(gx, 4), blk, 0, stream>>>(wA[10], 512, fuse1, 512, wb[10], nullptr, 0,
                                           d_out, NPTS, 512, ACT_LRELU, OUT_CM_F32);
}
